// BiaffineLinker_80977313399632
// MI455X (gfx1250) — matrix operand order repaired and checked by reading, not compiled, not run on hardware
//
#include <hip/hip_runtime.h>
#include <math.h>

// Problem constants (from reference)
#define BB_  16
#define SS_  2048
#define HH_  1024
#define NK_  256
#define NV_  512
#define EPSF 1e-8f

// ---------------------------------------------------------------------------
// Vector types for WMMA (CDNA5 wave32)
// ---------------------------------------------------------------------------
typedef __attribute__((ext_vector_type(16))) __bf16          v16bf;
typedef __attribute__((ext_vector_type(8)))  float           v8f;
typedef __attribute__((ext_vector_type(8)))  unsigned short  v8u;
typedef __attribute__((ext_vector_type(16))) unsigned short  v16u;

__device__ __forceinline__ unsigned short f2bf(float f) {
    unsigned int u = __float_as_uint(f);
    unsigned int r = u + 0x7FFFu + ((u >> 16) & 1u);   // round-to-nearest-even
    return (unsigned short)(r >> 16);
}

__device__ __forceinline__ v8f wmma_bf16(v16bf a, v16bf b, v8f c) {
    // (neg_a, A, neg_b, B, c_mod, C, reuse_a, reuse_b)
    return __builtin_amdgcn_wmma_f32_16x16x32_bf16(false, a, false, b,
                                                   (short)0, c, false, false);
}

// ---------------------------------------------------------------------------
// Tile loaders: contiguous k per lane half, same order in A and B (not the ISA table order)
// A 16x32 (MxK), row-major source (stride lda):
//   lanes 0-15:  row M=lane,    K = {0..15}
//   lanes 16-31: row M=lane-16, K = {16..31}
// -> two aligned 16B loads per lane.
// ---------------------------------------------------------------------------
__device__ __forceinline__ v16bf load_a_tile(const unsigned short* __restrict__ A,
                                             int lda, int lane) {
    const int m  = lane & 15;
    const int kc = (lane >> 4) << 4;             // 0 or 16
    const unsigned short* p = A + (long long)m * lda + kc;
    v8u lo = *(const v8u*)(p);
    v8u hi = *(const v8u*)(p + 8);
    v16u t;
#pragma unroll
    for (int i = 0; i < 8; ++i) { t[i] = lo[i]; t[8 + i] = hi[i]; }
    return __builtin_bit_cast(v16bf, t);
}

// B 32x16 (KxN) where the source is V row-major NxK (stride ldk), B = V^T:
//   lane holds column N=lane&15 of B = row n of V; K chunk {0..15} or {16..31}
// -> one aligned 32B load per lane (fully contiguous).
__device__ __forceinline__ v16bf load_b_tile_nt(const unsigned short* __restrict__ V,
                                                int ldk, int lane) {
    const int n  = lane & 15;
    const int kb = (lane >> 4) << 4;             // 0 or 16
    return __builtin_bit_cast(v16bf, *(const v16u*)(V + (long long)n * ldk + kb));
}

// ---------------------------------------------------------------------------
// K0: weights f32 -> bf16, TRANSPOSED (so GEMM B-loads are contiguous).
// Tiled 32x32 transpose through LDS; grid.z selects the matrix.
// ---------------------------------------------------------------------------
__global__ void transpose_weights_kernel(const float* __restrict__ W0,
                                         const float* __restrict__ W1,
                                         const float* __restrict__ W2,
                                         unsigned short* __restrict__ T0,
                                         unsigned short* __restrict__ T1,
                                         unsigned short* __restrict__ T2) {
    __shared__ unsigned short tile[32][33];      // +1 pad vs bank conflicts
    const float* W = (blockIdx.z == 0) ? W0 : (blockIdx.z == 1) ? W1 : W2;
    unsigned short* T = (blockIdx.z == 0) ? T0 : (blockIdx.z == 1) ? T1 : T2;
    const int x0 = blockIdx.x * 32;
    const int y0 = blockIdx.y * 32;
#pragma unroll
    for (int r = threadIdx.y; r < 32; r += 8)
        tile[r][threadIdx.x] = f2bf(W[(long long)(y0 + r) * HH_ + x0 + threadIdx.x]);
    __syncthreads();
#pragma unroll
    for (int r = threadIdx.y; r < 32; r += 8)
        T[(long long)(x0 + r) * HH_ + y0 + threadIdx.x] = tile[threadIdx.x][r];
}

// ---------------------------------------------------------------------------
// K1: gather tokens + f32 -> bf16   (one block per gathered row)
// ---------------------------------------------------------------------------
__global__ void gather_tokens_kernel(const float* __restrict__ seq,
                                     const int* __restrict__ kidx,
                                     const int* __restrict__ vidx,
                                     unsigned short* __restrict__ keyA,
                                     unsigned short* __restrict__ valA) {
    const int row = blockIdx.x;
    const int b   = row / (NK_ + NV_);
    const int r   = row % (NK_ + NV_);
    const float* src;
    unsigned short* dst;
    if (r < NK_) {
        const int t = kidx[b * NK_ + r];
        src = seq  + ((long long)b * SS_ + t) * HH_;
        dst = keyA + ((long long)b * NK_ + r) * HH_;
    } else {
        const int rv = r - NK_;
        const int t  = vidx[b * NV_ + rv];
        src = seq  + ((long long)b * SS_ + t)  * HH_;
        dst = valA + ((long long)b * NV_ + rv) * HH_;
    }
    for (int i = threadIdx.x; i < HH_; i += blockDim.x) dst[i] = f2bf(src[i]);
}

// ---------------------------------------------------------------------------
// K2: bf16 WMMA GEMM,  C[b] = A[b] (MxK, row-major) x V[b]^T (V: NxK row-major)
// Block = 256 threads = 8 waves arranged 4(M) x 2(N); block tile 128x128.
// Wave tile 32x64 = 2x4 WMMA 16x16x32 accumulators -> 8 WMMAs per k-step.
// KDIM is compile-time so the k-loop unrolls/pipelines.
// ---------------------------------------------------------------------------
template <int COUT_BF16, int KDIM>
__global__ void wmma_gemm_bf16_kernel(const unsigned short* __restrict__ A,
                                      long long strideA,
                                      const unsigned short* __restrict__ V,
                                      long long strideB, int ldb,
                                      const float* __restrict__ biasN,
                                      const float* __restrict__ biasScalar,
                                      unsigned short* __restrict__ Cbf,
                                      float* __restrict__ Cf,
                                      long long strideC, int ldc) {
    const int lane = threadIdx.x & 31;
    const int wave = threadIdx.x >> 5;
    const int wm   = wave >> 1;                       // 0..3
    const int wn   = wave & 1;                        // 0..1
    const int b    = blockIdx.z;
    const int m0   = blockIdx.y * 128 + wm * 32;
    const int n0   = blockIdx.x * 128 + wn * 64;

    const unsigned short* Ab = A + (long long)b * strideA;
    const unsigned short* Vb = V + (long long)b * strideB;
    const unsigned short* pa0 = Ab + (long long)m0 * KDIM;       // wave's A rows
    const unsigned short* pb0 = Vb + (long long)n0 * ldb;        // wave's V rows

    v8f acc[2][4];
#pragma unroll
    for (int mi = 0; mi < 2; ++mi)
#pragma unroll
        for (int ni = 0; ni < 4; ++ni)
#pragma unroll
            for (int r = 0; r < 8; ++r) acc[mi][ni][r] = 0.0f;

#pragma unroll 2
    for (int kt = 0; kt < KDIM; kt += 32) {
        const v16bf a0 = load_a_tile(pa0 + kt,            KDIM, lane);
        const v16bf a1 = load_a_tile(pa0 + 16 * KDIM + kt, KDIM, lane);
        v16bf bt[4];
#pragma unroll
        for (int ni = 0; ni < 4; ++ni)
            bt[ni] = load_b_tile_nt(pb0 + (long long)(16 * ni) * ldb + kt, ldb, lane);

        // Pull the next k-chunk toward the WGP ahead of use (global_prefetch_b8)
        if (kt + 256 < KDIM) {
            __builtin_prefetch(pa0 + (lane & 15) * KDIM + kt + 256, 0, 1);
            __builtin_prefetch(pb0 + (lane & 15) * ldb  + kt + 256, 0, 1);
        }

#pragma unroll
        for (int ni = 0; ni < 4; ++ni) {
            acc[0][ni] = wmma_bf16(a0, bt[ni], acc[0][ni]);
            acc[1][ni] = wmma_bf16(a1, bt[ni], acc[1][ni]);
        }
    }

    // Epilogue: 16x16 f32 C layout (VGPR r: lanes 0-15 -> M=r, lanes 16-31 -> M=8+r)
    const float bsc  = biasScalar ? *biasScalar : 0.0f;
    const int cn_off = lane & 15;
    const int cm_off = (lane >> 4) << 3;
#pragma unroll
    for (int mi = 0; mi < 2; ++mi)
#pragma unroll
        for (int ni = 0; ni < 4; ++ni) {
            const int cm = m0 + mi * 16 + cm_off;
            const int cn = n0 + ni * 16 + cn_off;
            const float bn = biasN ? biasN[cn] : 0.0f;
#pragma unroll
            for (int r = 0; r < 8; ++r) {
                const float v = acc[mi][ni][r] + bn + bsc;
                const long long off = (long long)b * strideC
                                    + (long long)(cm + r) * ldc + cn;
                if (COUT_BF16) Cbf[off] = f2bf(v);
                else           Cf[off]  = v;
            }
        }
}

// ---------------------------------------------------------------------------
// K3: per-pair spatial features + MLPs (weights cached in LDS)
// ---------------------------------------------------------------------------
__global__ void link_scores_kernel(const float* __restrict__ bbox,
                                   const int* __restrict__ kidx,
                                   const int* __restrict__ vidx,
                                   const float* __restrict__ biaff,
                                   const float* __restrict__ Ws1, const float* __restrict__ bs1,
                                   const float* __restrict__ Ws2, const float* __restrict__ bs2,
                                   const float* __restrict__ Wf1, const float* __restrict__ bf1,
                                   const float* __restrict__ Wf2, const float* __restrict__ bf2,
                                   float* __restrict__ out) {
    __shared__ float sWs1[8 * 64], sbs1[64];
    __shared__ float sWs2[64 * 32], sbs2[32];
    __shared__ float sWf1[33 * 16], sbf1[16], sWf2[16];

    for (int i = threadIdx.x; i < 8 * 64;  i += blockDim.x) sWs1[i] = Ws1[i];
    for (int i = threadIdx.x; i < 64;      i += blockDim.x) sbs1[i] = bs1[i];
    for (int i = threadIdx.x; i < 64 * 32; i += blockDim.x) sWs2[i] = Ws2[i];
    for (int i = threadIdx.x; i < 32;      i += blockDim.x) sbs2[i] = bs2[i];
    for (int i = threadIdx.x; i < 33 * 16; i += blockDim.x) sWf1[i] = Wf1[i];
    for (int i = threadIdx.x; i < 16;      i += blockDim.x) sbf1[i] = bf1[i];
    for (int i = threadIdx.x; i < 16;      i += blockDim.x) sWf2[i] = Wf2[i];
    __syncthreads();

    const long long idx   = (long long)blockIdx.x * blockDim.x + threadIdx.x;
    const long long total = (long long)BB_ * NK_ * NV_;
    if (idx >= total) return;

    const int v = (int)(idx % NV_);
    const long long t1 = idx / NV_;
    const int k = (int)(t1 % NK_);
    const int b = (int)(t1 / NK_);

    const float* kb = bbox + ((long long)b * SS_ + kidx[b * NK_ + k]) * 4;
    const float* vb = bbox + ((long long)b * SS_ + vidx[b * NV_ + v]) * 4;
    const float k0 = kb[0], k1 = kb[1], k2 = kb[2], k3 = kb[3];
    const float v0 = vb[0], v1 = vb[1], v2 = vb[2], v3 = vb[3];

    const float kcx = (k0 + k2) * 0.5f, kcy = (k1 + k3) * 0.5f;
    const float vcx = (v0 + v2) * 0.5f, vcy = (v1 + v3) * 0.5f;
    const float dx = vcx - kcx, dy = vcy - kcy;
    const float dist  = sqrtf(dx * dx + dy * dy + EPSF);
    const float angle = atan2f(dy, dx);
    const float kh = k3 - k1, vh = v3 - v1;
    const float ho = fmaxf(fminf(k3, v3) - fmaxf(k1, v1), 0.0f);
    const float halign = ho / (fminf(kh, vh) + EPSF);
    const float kw = k2 - k0, vw = v2 - v0;
    const float vo = fmaxf(fminf(k2, v2) - fmaxf(k0, v0), 0.0f);
    const float valign = vo / (fminf(kw, vw) + EPSF);
    const float aratio = (vh * vw) / (kh * kw + EPSF);
    const float aspect = (vw / (vh + EPSF)) / (kw / (kh + EPSF));
    const float f[8] = {dx, dy, dist, angle, halign, valign, aratio, aspect};

    float h1[64];
#pragma unroll 8
    for (int j = 0; j < 64; ++j) {
        float s = sbs1[j];
#pragma unroll
        for (int i = 0; i < 8; ++i) s += f[i] * sWs1[i * 64 + j];
        h1[j] = fmaxf(s, 0.0f);
    }
    float sp[32];
#pragma unroll 4
    for (int j = 0; j < 32; ++j) {
        float s = sbs2[j];
#pragma unroll 8
        for (int i = 0; i < 64; ++i) s += h1[i] * sWs2[i * 32 + j];
        sp[j] = s;
    }
    const float bia = biaff[idx];
    float score = bf2[0];
#pragma unroll 4
    for (int j = 0; j < 16; ++j) {
        float s = sbf1[j] + bia * sWf1[j];           // row 0 of Wf1 = biaffine ch.
#pragma unroll 8
        for (int i = 0; i < 32; ++i) s += sp[i] * sWf1[(i + 1) * 16 + j];
        score += fmaxf(s, 0.0f) * sWf2[j];
    }
    out[idx] = score;
}

// ---------------------------------------------------------------------------
// Host launcher
// ---------------------------------------------------------------------------
extern "C" void kernel_launch(void* const* d_in, const int* in_sizes, int n_in,
                              void* d_out, int out_size, void* d_ws, size_t ws_size,
                              hipStream_t stream) {
    (void)in_sizes; (void)n_in; (void)out_size; (void)ws_size;

    const float* seq  = (const float*)d_in[0];
    const float* bbox = (const float*)d_in[1];
    const int*   kidx = (const int*)d_in[2];
    const int*   vidx = (const int*)d_in[3];
    const float* Wk   = (const float*)d_in[4];
    const float* bk   = (const float*)d_in[5];
    const float* Wv   = (const float*)d_in[6];
    const float* bv   = (const float*)d_in[7];
    const float* Wb   = (const float*)d_in[8];
    const float* bb   = (const float*)d_in[9];
    const float* Ws1  = (const float*)d_in[10];
    const float* bs1  = (const float*)d_in[11];
    const float* Ws2  = (const float*)d_in[12];
    const float* bs2  = (const float*)d_in[13];
    const float* Wf1  = (const float*)d_in[14];
    const float* bf1  = (const float*)d_in[15];
    const float* Wf2  = (const float*)d_in[16];
    const float* bf2  = (const float*)d_in[17];

    // Workspace carve-out (bf16 activations + transposed bf16 weights; 54 MB)
    char* ws = (char*)d_ws;
    size_t off = 0;
    auto carve = [&](size_t bytes) -> char* {
        char* p = ws + off;
        off += (bytes + 255) & ~(size_t)255;
        return p;
    };
    unsigned short* wkt  = (unsigned short*)carve((size_t)HH_ * HH_ * 2);       // Wk^T
    unsigned short* wvt  = (unsigned short*)carve((size_t)HH_ * HH_ * 2);       // Wv^T
    unsigned short* wbt  = (unsigned short*)carve((size_t)HH_ * HH_ * 2);       // Wb^T
    unsigned short* keyA = (unsigned short*)carve((size_t)BB_ * NK_ * HH_ * 2); // 8 MB
    unsigned short* valA = (unsigned short*)carve((size_t)BB_ * NV_ * HH_ * 2); // 16 MB
    unsigned short* keyR = (unsigned short*)carve((size_t)BB_ * NK_ * HH_ * 2); // 8 MB
    unsigned short* valR = (unsigned short*)carve((size_t)BB_ * NV_ * HH_ * 2); // 16 MB
    unsigned short* kWb  = valA;            // valA dead after G2 (8 MB needed of 16)
    float*          biaf = (float*)keyA;    // keyA dead after G1 (exactly 8 MB)

    const dim3 blk(256);

    // K0: weights -> bf16, transposed (1024/32 = 32 tiles per side, 3 matrices)
    {
        dim3 g(HH_ / 32, HH_ / 32, 3);
        dim3 tb(32, 8);
        transpose_weights_kernel<<<g, tb, 0, stream>>>(Wk, Wv, Wb, wkt, wvt, wbt);
    }

    // K1: gather tokens -> bf16
    gather_tokens_kernel<<<BB_ * (NK_ + NV_), blk, 0, stream>>>(
        seq, kidx, vidx, keyA, valA);

    // G1: keyR = keyA @ Wk + bk          (M=256, N=1024, K=1024, per batch)
    {
        dim3 g(HH_ / 128, NK_ / 128, BB_);
        wmma_gemm_bf16_kernel<1, HH_><<<g, blk, 0, stream>>>(
            keyA, (long long)NK_ * HH_, wkt, 0, HH_,
            bk, nullptr, keyR, nullptr, (long long)NK_ * HH_, HH_);
    }
    // G2: valR = valA @ Wv + bv          (M=512)
    {
        dim3 g(HH_ / 128, NV_ / 128, BB_);
        wmma_gemm_bf16_kernel<1, HH_><<<g, blk, 0, stream>>>(
            valA, (long long)NV_ * HH_, wvt, 0, HH_,
            bv, nullptr, valR, nullptr, (long long)NV_ * HH_, HH_);
    }
    // G3: kWb = keyR @ Wb                (no bias)
    {
        dim3 g(HH_ / 128, NK_ / 128, BB_);
        wmma_gemm_bf16_kernel<1, HH_><<<g, blk, 0, stream>>>(
            keyR, (long long)NK_ * HH_, wbt, 0, HH_,
            nullptr, nullptr, kWb, nullptr, (long long)NK_ * HH_, HH_);
    }
    // G4: biaffine = kWb @ valR^T + bb   (M=256, N=512, K=1024, f32 out)
    {
        dim3 g(NV_ / 128, NK_ / 128, BB_);
        wmma_gemm_bf16_kernel<0, HH_><<<g, blk, 0, stream>>>(
            kWb, (long long)NK_ * HH_, valR, (long long)NV_ * HH_, HH_,
            nullptr, bb, nullptr, biaf, (long long)NK_ * NV_, NV_);
    }
    // K3: spatial features + MLPs -> link scores
    {
        const long long total = (long long)BB_ * NK_ * NV_;
        link_scores_kernel<<<(unsigned)((total + 255) / 256), blk, 0, stream>>>(
            bbox, kidx, vidx, biaf,
            Ws1, bs1, Ws2, bs2, Wf1, bf1, Wf2, bf2, (float*)d_out);
    }
}
